// SmallEmbedFFNN_2327872274768
// MI455X (gfx1250) — matrix operand order repaired and checked by reading, not compiled, not run on hardware
//
#include <hip/hip_runtime.h>
#include <hip/hip_bf16.h>
#include <math.h>

// Problem constants (from reference)
#define BB 1024
#define PP 64
#define EE 32
#define VV 1000
#define NTT 33
#define D1 1120   // 35*E input features
#define H1 512
#define H2 256

typedef _Float16 v16h __attribute__((ext_vector_type(16)));
typedef _Float16 h8   __attribute__((ext_vector_type(8)));
typedef _Float16 h4   __attribute__((ext_vector_type(4)));
typedef float    v8f  __attribute__((ext_vector_type(8)));

// LDS layout (bytes)
#define FST   1128            // feats row stride in halves (padded: 564 dwords, gcd(52,64)=4 -> conflict free for 16 lanes)
#define H1ST  520             // h1 row stride in halves
#define H2ST  263             // h2 row stride in floats (odd dword stride -> conflict free)
#define SM_FEATS 0
#define SM_H1   (64 * FST * 2)                 // 144384
#define SM_H2   (SM_H1 + 64 * H1ST * 2)        // 210944
#define SM_LG   (SM_H2 + 64 * H2ST * 4)        // 278272
#define SM_TOTAL (SM_LG + 64 * 4)              // 278528  (< 320KB WGP LDS)

// Packed weight sizes (halves)
#define PW1_HALVES (35 * 32 * 32 * 16)   // 573440  (35 kTiles x 32 nTiles x 32 lanes x 16)
#define PW2_HALVES (16 * 16 * 32 * 16)   // 131072

// ---------------------------------------------------------------------------
// Pack W1 (1120x512 f32, row-major K x N) into WMMA B-tile lane layout, f16.
// Tile (kt, nt): lane = 16*lh + lr holds halves i=0..15 with
//   K = kt*32 + (i < 8 ? lh*8 + i : 16 + lh*8 + (i - 8)),  N = nt*16 + lr
// flat p = kt*16384 + nt*512 + lane*16 + i
// ---------------------------------------------------------------------------
__global__ void pack_w1_kernel(const float* __restrict__ W1, _Float16* __restrict__ out) {
    int p = blockIdx.x * blockDim.x + threadIdx.x;
    if (p >= PW1_HALVES) return;
    int i    = p & 15;
    int lane = (p >> 4) & 31;
    int nt   = (p >> 9) & 31;
    int kt   = p >> 14;
    int lh = lane >> 4, lr = lane & 15;
    int K = kt * 32 + (i < 8 ? lh * 8 + i : 16 + lh * 8 + (i - 8));
    int N = nt * 16 + lr;
    out[p] = (_Float16)W1[K * H1 + N];
}

__global__ void pack_w2_kernel(const float* __restrict__ W2, _Float16* __restrict__ out) {
    int p = blockIdx.x * blockDim.x + threadIdx.x;
    if (p >= PW2_HALVES) return;
    int i    = p & 15;
    int lane = (p >> 4) & 31;
    int nt   = (p >> 9) & 15;
    int kt   = p >> 13;
    int lh = lane >> 4, lr = lane & 15;
    int K = kt * 32 + (i < 8 ? lh * 8 + i : 16 + lh * 8 + (i - 8));
    int N = nt * 16 + lr;
    out[p] = (_Float16)W2[K * H2 + N];
}

// ---------------------------------------------------------------------------
// Main fused kernel: one workgroup (8 wave32) per batch element.
//   Stage 0: gather features -> f16 LDS (64 x 1120)
//   Stage 1: feats @ W1 + b1, relu -> h1 LDS (f16) via v_wmma_f32_16x16x32_f16
//   Stage 2: h1 @ W2 + b2, relu -> h2 LDS (f32)
//   Stage 3: h2 @ W3 + b3 -> logits; softmax over the 64 rows -> out
// ---------------------------------------------------------------------------
__global__ void __launch_bounds__(256, 1)
embed_ffnn_softmax_kernel(const float* __restrict__ x_float,
                          const int*   __restrict__ x_idx,
                          const float* __restrict__ emb,
                          const float* __restrict__ b1,
                          const float* __restrict__ b2,
                          const float* __restrict__ W3,
                          const float* __restrict__ b3,
                          const _Float16* __restrict__ pW1,
                          const _Float16* __restrict__ pW2,
                          float* __restrict__ out) {
    extern __shared__ char smem[];
    _Float16* feats = (_Float16*)(smem + SM_FEATS);
    _Float16* h1s   = (_Float16*)(smem + SM_H1);
    float*    h2s   = (float*)(smem + SM_H2);
    float*    lgts  = (float*)(smem + SM_LG);

    const int b    = blockIdx.x;
    const int tid  = threadIdx.x;
    const int w    = tid >> 5;      // wave id 0..7
    const int lane = tid & 31;
    const int lh   = lane >> 4;     // lane half
    const int lr   = lane & 15;     // lane row/col within half

    // ---------------- Stage 0: build feats (f16) in LDS -------------------
    // 64 rows x 1120 cols = 17920 float4 chunks; 70 per thread.
    for (int it = 0; it < 70; ++it) {
        int flat = tid + it * 256;          // 0..17919
        int r = flat / 280;
        int q = flat - r * 280;
        int c = q << 2;                     // feature column (multiple of 4)
        float4 v;
        if (c < 64) {
            float x = x_float[(b * PP + r) * 2 + (c >> 5)];
            v = make_float4(x, x, x, x);
        } else {
            int cc = c - 64;
            int t  = cc >> 5;
            int e  = cc & 31;
            int idx = x_idx[(b * PP + r) * NTT + t];
            v = *(const float4*)(emb + ((long)t * VV + idx) * EE + e);
        }
        h4 hv = { (_Float16)v.x, (_Float16)v.y, (_Float16)v.z, (_Float16)v.w };
        *(h4*)(feats + r * FST + c) = hv;
    }
    __syncthreads();

    // ---------------- Stage 1: layer 1 GEMM (64x1120 @ 1120x512) ----------
    // Wave w: nTiles {4w..4w+3}, all 4 mTiles. 16 accumulators.
    {
        v8f acc[4][4];
#pragma unroll
        for (int m = 0; m < 4; ++m)
#pragma unroll
            for (int n = 0; n < 4; ++n)
                acc[m][n] = (v8f){0.f, 0.f, 0.f, 0.f, 0.f, 0.f, 0.f, 0.f};

        for (int k = 0; k < 35; ++k) {
            v16h A[4];
#pragma unroll
            for (int m = 0; m < 4; ++m) {
                const _Float16* p0 = feats + (m * 16 + lr) * FST + k * 32 + 8 * lh;
                h8 lo = *(const h8*)p0;          // K = kbase + 8*lh + 0..7
                h8 hi = *(const h8*)(p0 + 16);   // K = kbase + 16 + 8*lh + 0..7
#pragma unroll
                for (int i = 0; i < 8; ++i) { A[m][i] = lo[i]; A[m][i + 8] = hi[i]; }
            }
#pragma unroll
            for (int n = 0; n < 4; ++n) {
                int nt = w * 4 + n;
                v16h Bv = *(const v16h*)(pW1 + (((k * 32 + nt) * 32 + lane) << 4));
#pragma unroll
                for (int m = 0; m < 4; ++m)
                    acc[m][n] = __builtin_amdgcn_wmma_f32_16x16x32_f16(
                        false, A[m], false, Bv, (short)0, acc[m][n], false, false);
            }
        }
        // Epilogue: + b1, relu, f16 -> h1 LDS (row-major)
#pragma unroll
        for (int n = 0; n < 4; ++n) {
            int col = (w * 4 + n) * 16 + lr;
            float bias = b1[col];
#pragma unroll
            for (int m = 0; m < 4; ++m)
#pragma unroll
                for (int j = 0; j < 8; ++j) {
                    float val = fmaxf(acc[m][n][j] + bias, 0.f);
                    int row = m * 16 + j + 8 * lh;
                    h1s[row * H1ST + col] = (_Float16)val;
                }
        }
    }
    __syncthreads();

    // ---------------- Stage 2: layer 2 GEMM (64x512 @ 512x256) ------------
    // Wave w: nTiles {2w, 2w+1}, all 4 mTiles.
    {
        v8f acc[4][2];
#pragma unroll
        for (int m = 0; m < 4; ++m)
#pragma unroll
            for (int n = 0; n < 2; ++n)
                acc[m][n] = (v8f){0.f, 0.f, 0.f, 0.f, 0.f, 0.f, 0.f, 0.f};

        for (int k = 0; k < 16; ++k) {
            v16h A[4];
#pragma unroll
            for (int m = 0; m < 4; ++m) {
                const _Float16* p0 = h1s + (m * 16 + lr) * H1ST + k * 32 + 8 * lh;
                h8 lo = *(const h8*)p0;
                h8 hi = *(const h8*)(p0 + 16);
#pragma unroll
                for (int i = 0; i < 8; ++i) { A[m][i] = lo[i]; A[m][i + 8] = hi[i]; }
            }
#pragma unroll
            for (int n = 0; n < 2; ++n) {
                int nt = w * 2 + n;
                v16h Bv = *(const v16h*)(pW2 + (((k * 16 + nt) * 32 + lane) << 4));
#pragma unroll
                for (int m = 0; m < 4; ++m)
                    acc[m][n] = __builtin_amdgcn_wmma_f32_16x16x32_f16(
                        false, A[m], false, Bv, (short)0, acc[m][n], false, false);
            }
        }
        // Epilogue: + b2, relu, f32 -> h2 LDS
#pragma unroll
        for (int n = 0; n < 2; ++n) {
            int col = (w * 2 + n) * 16 + lr;
            float bias = b2[col];
#pragma unroll
            for (int m = 0; m < 4; ++m)
#pragma unroll
                for (int j = 0; j < 8; ++j) {
                    float val = fmaxf(acc[m][n][j] + bias, 0.f);
                    int row = m * 16 + j + 8 * lh;
                    h2s[row * H2ST + col] = val;
                }
        }
    }
    __syncthreads();

    // ---------------- Stage 3: layer 3 dot + softmax over P ---------------
    if (tid < 64) {
        float acc = b3[0];
        for (int c = 0; c < H2; ++c)
            acc += h2s[tid * H2ST + c] * W3[c];
        lgts[tid] = acc;
    }
    __syncthreads();
    if (tid < 64) {
        float mx = -INFINITY;
        for (int i = 0; i < 64; ++i) mx = fmaxf(mx, lgts[i]);
        float s = 0.f;
        for (int i = 0; i < 64; ++i) s += __expf(lgts[i] - mx);
        out[b * PP + tid] = __expf(lgts[tid] - mx) / s;
    }
}

// ---------------------------------------------------------------------------
extern "C" void kernel_launch(void* const* d_in, const int* in_sizes, int n_in,
                              void* d_out, int out_size, void* d_ws, size_t ws_size,
                              hipStream_t stream) {
    const float* x_float = (const float*)d_in[0];
    const int*   x_idx   = (const int*)d_in[1];
    const float* emb     = (const float*)d_in[2];
    const float* W1      = (const float*)d_in[3];
    const float* b1      = (const float*)d_in[4];
    const float* W2      = (const float*)d_in[5];
    const float* b2      = (const float*)d_in[6];
    const float* W3      = (const float*)d_in[7];
    const float* b3      = (const float*)d_in[8];
    float* out = (float*)d_out;

    _Float16* pW1 = (_Float16*)d_ws;                                   // 1,146,880 B
    _Float16* pW2 = (_Float16*)((char*)d_ws + (size_t)PW1_HALVES * 2); // +262,144 B

    pack_w1_kernel<<<(PW1_HALVES + 255) / 256, 256, 0, stream>>>(W1, pW1);
    pack_w2_kernel<<<(PW2_HALVES + 255) / 256, 256, 0, stream>>>(W2, pW2);

    embed_ffnn_softmax_kernel<<<BB, 256, SM_TOTAL, stream>>>(
        x_float, x_idx, emb, b1, b2, W3, b3, pW1, pW2, out);
}
